// SequenceCriteriaCRF_20899310862338
// MI455X (gfx1250) — matrix operand order repaired and checked by reading, not compiled, not run on hardware
//
#include <hip/hip_runtime.h>
#include <hip/hip_bf16.h>
#include <type_traits>

typedef __attribute__((ext_vector_type(16))) _Float16 v16h;
typedef __attribute__((ext_vector_type(8)))  _Float16 h8;
typedef __attribute__((ext_vector_type(8)))  float    v8f;
typedef __attribute__((ext_vector_type(4)))  float    f4;

#define Bn 512
#define Tn 1024
#define Kn 64
#define ROWS 16   // batch rows per block (WMMA M)

#define LOG2E 1.4426950408889634f
#define LN2   0.6931471805599453f

// raw v_exp_f32 / v_log_f32 (no denorm fixup path)
__device__ __forceinline__ float fast_exp(float x) {
  return __builtin_amdgcn_exp2f(x * LOG2E);
}
__device__ __forceinline__ float fast_log(float x) {
  return __builtin_amdgcn_logf(x) * LN2;
}

// ---------------------------------------------------------------------------
// Forward algorithm: exp-space scan. alpha tile (16 x 64) lives in LDS,
// exp(transitions) B-operands live in VGPRs for the whole T loop.
// Each of the 4 waves owns a 16x16 output tile (16 batch rows x 16 labels).
// ---------------------------------------------------------------------------
__global__ __launch_bounds__(128) void crf_forward_kernel(
    const float* __restrict__ emissions,
    const float* __restrict__ transitions,
    const float* __restrict__ start_trans,
    const float* __restrict__ end_trans,
    const int*   __restrict__ sent_len,
    float* __restrict__ fwd_sum)
{
  __shared__ __align__(16) float    alpha[ROWS][Kn];  // running log-alphas
  __shared__ __align__(16) _Float16 aexp [ROWS][Kn];  // exp(alpha-max), WMMA A
  __shared__ __align__(16) _Float16 expT [Kn][Kn];    // exp(transitions), WMMA B
  __shared__ __align__(16) float    mrow [ROWS];      // per-row shift

  const int tid   = threadIdx.x;
  const int wave  = tid >> 5;               // 0..3  -> 16-column slice
  const int lane  = tid & 31;
  const int row   = tid >> 3;               // 0..15 -> alpha row (8 thr/row)
  const int sub   = tid & 7;                // 0..7  -> 8 cols each
  const int tileb = blockIdx.x * ROWS;      // first batch row of this block

  const int hi = lane >> 4;                 // lane half-select
  const int nl = lane & 15;
  const int j0 = wave * 16;
  const int j  = j0 + nl;                   // this lane's output column

  // exp(transitions) -> LDS (f16)
  for (int idx = tid; idx < Kn * Kn; idx += 128)
    ((_Float16*)expT)[idx] = (_Float16)fast_exp(transitions[idx]);

  // alpha0 = start_trans + emissions[:,0,:]
  for (int idx = tid; idx < ROWS * Kn; idx += 128) {
    int m = idx >> 6, jj = idx & (Kn - 1);
    alpha[m][jj] = start_trans[jj] +
                   emissions[((size_t)(tileb + m)) * Tn * Kn + jj];
  }

  // per-lane constants for the write-back rows (M = v + 8*hi)
  int slen_r[8];
  int rbase [8];                            // emissions row base + column
#pragma unroll
  for (int v = 0; v < 8; ++v) {
    int b = tileb + 8 * hi + v;
    slen_r[v] = sent_len[b];
    rbase [v] = b * (Tn * Kn) + j;
  }
  __syncthreads();

  // Current alpha values of the 8 elements this lane owns (C/D layout).
  float aprev[8];
#pragma unroll
  for (int v = 0; v < 8; ++v) aprev[v] = alpha[v + 8 * hi][j];

  // Tile-uniform minimum sentence length: below it no masking is needed.
  int tmin = slen_r[0];
#pragma unroll
  for (int v = 1; v < 8; ++v) tmin = min(tmin, slen_r[v]);
  tmin = min(tmin, __shfl_xor(tmin, 16));   // combine the two lane halves
  tmin = __builtin_amdgcn_readfirstlane(tmin);
  tmin = max(tmin, 1);

  // Pre-pack this wave's B operands (constant across all timesteps).
  // B 32x16 f16, this code's contiguous k order: lanes 0-15 hold K=0..15 (reg r -> K=2r,2r+1),
  // lanes 16-31 hold K=16..31.
  v16h b0, b1;
#pragma unroll
  for (int r = 0; r < 8; ++r) {
    int k = hi * 16 + 2 * r;
    b0[2 * r]     = expT[k     ][j];
    b0[2 * r + 1] = expT[k + 1 ][j];
    b1[2 * r]     = expT[k + 32][j];
    b1[2 * r + 1] = expT[k + 33][j];
  }

  auto step = [&](int t, auto masked_) {
    constexpr bool MASKED = decltype(masked_)::value;
    const int toff = t * Kn;

    // ---- issue this step's emission loads early (latency hidden) ----
    float ev[8];
#pragma unroll
    for (int v = 0; v < 8; ++v) ev[v] = emissions[rbase[v] + toff];
    __builtin_prefetch(&emissions[rbase[0] + toff + Kn], 0, 1);

    // ---- row max via 2x b128 LDS reads + 3 xor-shuffles (no barrier) ----
    f4 aA = *(const f4*)&alpha[row][sub * 8];
    f4 aB = *(const f4*)&alpha[row][sub * 8 + 4];
    float m = fmaxf(fmaxf(fmaxf(aA.x, aA.y), fmaxf(aA.z, aA.w)),
                    fmaxf(fmaxf(aB.x, aB.y), fmaxf(aB.z, aB.w)));
    m = fmaxf(m, __shfl_xor(m, 1));
    m = fmaxf(m, __shfl_xor(m, 2));
    m = fmaxf(m, __shfl_xor(m, 4));
    if (sub == 0) mrow[row] = m;

    // ---- exp-shift into f16 A tile (one b128 store) ----
    h8 hx;
    hx[0] = (_Float16)fast_exp(aA.x - m);
    hx[1] = (_Float16)fast_exp(aA.y - m);
    hx[2] = (_Float16)fast_exp(aA.z - m);
    hx[3] = (_Float16)fast_exp(aA.w - m);
    hx[4] = (_Float16)fast_exp(aB.x - m);
    hx[5] = (_Float16)fast_exp(aB.y - m);
    hx[6] = (_Float16)fast_exp(aB.z - m);
    hx[7] = (_Float16)fast_exp(aB.w - m);
    *(h8*)&aexp[row][sub * 8] = hx;
    __syncthreads();                        // barrier 1: aexp + mrow ready

    // ---- pack A (16x32 f16, same contiguous k order as B: reg r -> K=hi*16+2r) ----
    v16h a0, a1;
#pragma unroll
    for (int r = 0; r < 8; ++r) {
      int k = hi * 16 + 2 * r;
      a0[2 * r]     = aexp[nl][k];
      a0[2 * r + 1] = aexp[nl][k + 1];
      a1[2 * r]     = aexp[nl][k + 32];
      a1[2 * r + 1] = aexp[nl][k + 33];
    }

    // ---- (16x64) @ (64x16) in exp space: two chained WMMAs over K ----
    v8f c = {};
    c = __builtin_amdgcn_wmma_f32_16x16x32_f16(false, a0, false, b0,
                                               (short)0, c, false, false);
    c = __builtin_amdgcn_wmma_f32_16x16x32_f16(false, a1, false, b1,
                                               (short)0, c, false, false);

    // ---- write-back: alpha' = log(C) + rowmax + emit ----
    // C/D layout: lane -> N, reg v -> M = v + 8*hi
    f4 mA = *(const f4*)&mrow[8 * hi];
    f4 mB = *(const f4*)&mrow[8 * hi + 4];
    float mv[8] = {mA.x, mA.y, mA.z, mA.w, mB.x, mB.y, mB.z, mB.w};
#pragma unroll
    for (int v = 0; v < 8; ++v) {
      float nv = fast_log(c[v]) + mv[v] + ev[v];
      if constexpr (MASKED) nv = (t < slen_r[v]) ? nv : aprev[v];
      aprev[v] = nv;
      alpha[v + 8 * hi][j] = nv;
    }
    __syncthreads();                        // barrier 2: alpha updated
  };

  // Unmasked scan while every row in the tile is still active, then masked.
  for (int t = 1; t < tmin; ++t)
    step(t, std::integral_constant<bool, false>{});
  for (int t = tmin; t < Tn; ++t)
    step(t, std::integral_constant<bool, true>{});

  // ---- fwd[b] = logsumexp_j(alpha[b][j] + end_trans[j]) ----
  float vals[8];
#pragma unroll
  for (int i = 0; i < 8; ++i)
    vals[i] = alpha[row][sub * 8 + i] + end_trans[sub * 8 + i];
  float m2 = vals[0];
#pragma unroll
  for (int i = 1; i < 8; ++i) m2 = fmaxf(m2, vals[i]);
  m2 = fmaxf(m2, __shfl_xor(m2, 1));
  m2 = fmaxf(m2, __shfl_xor(m2, 2));
  m2 = fmaxf(m2, __shfl_xor(m2, 4));
  float s = 0.f;
#pragma unroll
  for (int i = 0; i < 8; ++i) s += fast_exp(vals[i] - m2);
  s += __shfl_xor(s, 1);
  s += __shfl_xor(s, 2);
  s += __shfl_xor(s, 4);
  if (sub == 0) atomicAdd(fwd_sum, fast_log(s) + m2);
}

// ---------------------------------------------------------------------------
// Gold path score: gather + reduce, one block per batch element.
// ---------------------------------------------------------------------------
__global__ __launch_bounds__(128) void crf_gold_kernel(
    const float* __restrict__ emissions,
    const float* __restrict__ transitions,
    const float* __restrict__ start_trans,
    const float* __restrict__ end_trans,
    const int*   __restrict__ labels,
    const int*   __restrict__ sent_len,
    float* __restrict__ gold_sum)
{
  __shared__ float red[128];
  const int b   = blockIdx.x;
  const int len = sent_len[b];
  const int* lab = labels + (size_t)b * Tn;

  float s = 0.f;
  for (int t = threadIdx.x; t < len; t += 128) {
    int lt = lab[t];
    s += emissions[(((size_t)b) * Tn + t) * Kn + lt];
    if (t >= 1) s += transitions[lab[t - 1] * Kn + lt];
  }
  if (threadIdx.x == 0)
    s += start_trans[lab[0]] + end_trans[lab[len - 1]];

  red[threadIdx.x] = s;
  __syncthreads();
  for (int off = 64; off > 0; off >>= 1) {
    if (threadIdx.x < off) red[threadIdx.x] += red[threadIdx.x + off];
    __syncthreads();
  }
  if (threadIdx.x == 0) atomicAdd(gold_sum, red[0]);
}

__global__ void crf_init_kernel(float* acc) {
  acc[0] = 0.f;   // fwd sum
  acc[1] = 0.f;   // gold sum
}

__global__ void crf_finalize_kernel(const float* acc, float* out) {
  out[0] = (acc[0] - acc[1]) / (float)Bn;
}

extern "C" void kernel_launch(void* const* d_in, const int* in_sizes, int n_in,
                              void* d_out, int out_size, void* d_ws, size_t ws_size,
                              hipStream_t stream) {
  (void)in_sizes; (void)n_in; (void)out_size; (void)ws_size;
  const float* emissions   = (const float*)d_in[0];
  const float* transitions = (const float*)d_in[1];
  const float* start_trans = (const float*)d_in[2];
  const float* end_trans   = (const float*)d_in[3];
  const int*   labels      = (const int*)d_in[4];
  const int*   sent_len    = (const int*)d_in[5];
  float* out = (float*)d_out;
  float* acc = (float*)d_ws;   // [0]=fwd_sum, [1]=gold_sum

  crf_init_kernel<<<1, 1, 0, stream>>>(acc);
  crf_gold_kernel<<<Bn, 128, 0, stream>>>(emissions, transitions, start_trans,
                                          end_trans, labels, sent_len, acc + 1);
  crf_forward_kernel<<<Bn / ROWS, 128, 0, stream>>>(emissions, transitions,
                                                    start_trans, end_trans,
                                                    sent_len, acc);
  crf_finalize_kernel<<<1, 1, 0, stream>>>(acc, out);
}
